// SingleBidirectionalLSTMEncoder_79783312491110
// MI455X (gfx1250) — matrix operand order repaired and checked by reading, not compiled, not run on hardware
//
#include <hip/hip_runtime.h>

// ---------------- problem constants ----------------
#define B_      64
#define T_      512
#define EMBED_  256
#define UNITS_  512
#define G_      2048            // 4*UNITS
#define NBLK_PER_DIR  16
#define UNITS_PER_BLK 32        // 512 / 16
#define HC_OFF  (B_*T_*1024)    // start of final_state_h in d_out
#define HSZ_    (B_*1024)       // 65536

// ---------------- LDS layout (bytes) ----------------
// All row strides are multiples of 16B so tile fragments load as ds_load_b128.
// Ut : bf16 [128][520]  U-slice, transposed (n-major)    133120 B
// Wt : bf16 [128][264]  W-slice, transposed (n-major)     67584 B
// hs : bf16 [64][520]   current h (full 512 units)        66560 B  (filled by TDM)
// xs : bf16 [64][264]   embedding rows for step t         33792 B  (aliased by zs)
// zs : f32  [64][132]   gate pre-activations (alias xs)   33792 B
// msk: int  [64]                                            256 B
#define UT_OFF   0
#define UT_STR   520
#define WT_OFF   133120
#define WT_STR   264
#define HS_OFF   200704
#define HS_STR   520
#define XS_OFF   267264
#define XS_STR   264
#define ZS_STR   132
#define MSK_OFF  301056
#define SMEM_BYTES 301312

typedef __attribute__((ext_vector_type(16))) __bf16   v16bf;
typedef __attribute__((ext_vector_type(8)))  float    v8f;
typedef __attribute__((ext_vector_type(4)))  unsigned v4u;
typedef __attribute__((ext_vector_type(8)))  int      v8i;
typedef __attribute__((ext_vector_type(4)))  int      v4i;

union Tile { v4u q[2]; v16bf v; };       // 16 bf16 fragment as two 16B chunks

__device__ __forceinline__ float sigmoidf_(float x) {
  return 1.0f / (1.0f + __expf(-x));
}

__global__ void init_ws_kernel(unsigned* __restrict__ p, int n) {
  int i = blockIdx.x * blockDim.x + threadIdx.x;
  int stride = gridDim.x * blockDim.x;
  for (; i < n; i += stride) p[i] = 0u;
}

// Grid: 32 blocks = 2 directions x 16 unit-slices. 256 threads = 8 wave32.
__global__ __launch_bounds__(256, 1)
void lstm_bidir_kernel(const int*   __restrict__ enc,
                       const float* __restrict__ emb,
                       const float* __restrict__ Wf, const float* __restrict__ Uf,
                       const float* __restrict__ bfv,
                       const float* __restrict__ Wb, const float* __restrict__ Ub,
                       const float* __restrict__ bbv,
                       float* __restrict__ out,
                       int*   __restrict__ cnt,      // [2][T_] step barriers
                       __bf16* __restrict__ hbuf)    // [2 dir][2 parity][64*512]
{
  extern __shared__ char smem[];
  __bf16* Ut  = (__bf16*)(smem + UT_OFF);
  __bf16* Wt  = (__bf16*)(smem + WT_OFF);
  __bf16* hs  = (__bf16*)(smem + HS_OFF);
  __bf16* xs  = (__bf16*)(smem + XS_OFF);
  float*  zs  = (float*) (smem + XS_OFF);   // alias: valid after GEMM reads done
  int*    msk = (int*)   (smem + MSK_OFF);

  const int tid = threadIdx.x;
  const int dir = blockIdx.x / NBLK_PER_DIR;       // 0 = fwd, 1 = bwd
  const int blk = blockIdx.x % NBLK_PER_DIR;
  const int u0  = blk * UNITS_PER_BLK;

  const float* Wg = dir ? Wb : Wf;
  const float* Ug = dir ? Ub : Uf;
  const float* bg = dir ? bbv : bfv;
  int*    mycnt   = cnt  + dir * T_;
  __bf16* hb_base = hbuf + dir * 2 * (B_ * UNITS_);

  // ---- one-time: stage weight slices to LDS as bf16, transposed [n][k] ----
  // n_local -> global gate column: (n>>5)*512 + u0 + (n&31)
  for (int idx = tid; idx < 128 * EMBED_; idx += 256) {
    int n = idx & 127, k = idx >> 7;
    int g = ((n >> 5) * UNITS_) + u0 + (n & 31);
    Wt[n * WT_STR + k] = (__bf16)Wg[(size_t)k * G_ + g];
  }
  for (int idx = tid; idx < 128 * UNITS_; idx += 256) {
    int n = idx & 127, k = idx >> 7;
    int g = ((n >> 5) * UNITS_) + u0 + (n & 31);
    Ut[n * UT_STR + k] = (__bf16)Ug[(size_t)k * G_ + g];
  }

  // ---- per-thread state: thread owns unit ul for 8 batch rows ----
  const int ul  = tid & 31;        // unit within block's 32
  const int bgI = tid >> 5;        // batch group 0..7 -> b = bgI*8 + j
  const float bias_i = bg[0 * UNITS_ + u0 + ul];
  const float bias_f = bg[1 * UNITS_ + u0 + ul];
  const float bias_g = bg[2 * UNITS_ + u0 + ul];
  const float bias_o = bg[3 * UNITS_ + u0 + ul];
  float hreg[8], creg[8];
  #pragma unroll
  for (int j = 0; j < 8; ++j) { hreg[j] = 0.0f; creg[j] = 0.0f; }

  // WMMA lane decomposition (wave32)
  const int lane = tid & 31;
  const int l15  = lane & 15;
  const int kh   = lane >> 4;      // which K-half this lane carries
  const int wv   = tid >> 5;       // wave id 0..7 -> owns N-tile wv

  const unsigned hs_lds = (unsigned)(size_t)(void*)hs;   // LDS byte address of hs

  for (int s = 0; s < T_; ++s) {
    const int t  = dir ? (T_ - 1 - s) : s;
    const int pr = s & 1;          // read parity of h buffer
    const int pw = pr ^ 1;         // write parity

    // ---- TDM: DMA full h (64x512 bf16) global -> LDS with row padding ----
    // D# (ISA ch.8): 2D tile, data_size=8B, 128 units/row, 64 rows,
    // pad_enable, pad_interval=256 dwords (1 row), pad_amount=4 dwords (16B)
    // so LDS rows land at stride HS_STR = 520 bf16.
    if (tid < 32) {
      unsigned long long ga =
          (unsigned long long)(size_t)(hb_base + pr * (B_ * UNITS_));
      v4u g0;
      g0[0] = 1u;                                       // count=1, user mode
      g0[1] = hs_lds;                                   // lds_addr
      g0[2] = (unsigned)ga;                             // global_addr[31:0]
      g0[3] = ((unsigned)(ga >> 32) & 0x01FFFFFFu) | 0x80000000u; // addr hi | type=2
      v8i g1;
      g1[0] = 0x07D30000;            // data_size=8B, pad_en, interval=7, amount=3
      g1[1] = (int)(128u << 16);     // tensor_dim0 = 128 units (low16 @ bits63:48)
      g1[2] = (int)(64u << 16);      // dim0 hi16=0 | tensor_dim1 = 64
      g1[3] = (int)(128u << 16);     // tile_dim0 = 128 units
      g1[4] = 64;                    // tile_dim1 = 64 rows (tile_dim2 = 0)
      g1[5] = 128;                   // tensor_dim0_stride = 128 units
      g1[6] = 0; g1[7] = 0;
      v4i g2 = {0, 0, 0, 0};
      v4i g3 = {0, 0, 0, 0};
      v8i g4 = {0, 0, 0, 0, 0, 0, 0, 0};
      __builtin_amdgcn_tensor_load_to_lds(g0, g1, g2, g3, g4, 0);
      __builtin_amdgcn_s_wait_tensorcnt(0);
    }

    // ---- stage x_t (embedding gather, fp32->bf16) into LDS ----
    {
      int b = tid >> 2, q = tid & 3;
      int tok = enc[b * T_ + t];
      if (q == 0) msk[b] = (tok != 0);
      const float* er = emb + (size_t)tok * EMBED_ + q * 64;
      __bf16* xr = xs + b * XS_STR + q * 64;
      #pragma unroll 8
      for (int i = 0; i < 64; ++i) xr[i] = (__bf16)er[i];

      // prefetch next step's embedding row (global_prefetch_b8)
      int tn = dir ? (t - 1) : (t + 1);
      if (q == 0 && tn >= 0 && tn < T_) {
        int tok2 = enc[b * T_ + tn];
        __builtin_prefetch(emb + (size_t)tok2 * EMBED_, 0, 1);
      }
    }
    __syncthreads();

    // ---- z = x_t @ Wslice + h @ Uslice   (M=64, N=128, K=256+512) ----
    v8f acc[4] = {};

    #pragma unroll 2
    for (int kt = 0; kt < 8; ++kt) {            // x@W part, K = 256
      const int kb = kt * 32;
      Tile bt;
      const __bf16* bp = Wt + (wv * 16 + l15) * WT_STR + kb + kh * 8;
      bt.q[0] = *(const v4u*)(bp);
      bt.q[1] = *(const v4u*)(bp + 16);
      #pragma unroll
      for (int mt = 0; mt < 4; ++mt) {
        Tile at;
        const __bf16* ap = xs + (mt * 16 + l15) * XS_STR + kb + kh * 8;
        at.q[0] = *(const v4u*)(ap);
        at.q[1] = *(const v4u*)(ap + 16);
        acc[mt] = __builtin_amdgcn_wmma_f32_16x16x32_bf16(
            false, at.v, false, bt.v, (short)0, acc[mt], false, false);
      }
    }
    #pragma unroll 2
    for (int kt = 0; kt < 16; ++kt) {           // h@U part, K = 512
      const int kb = kt * 32;
      Tile bt;
      const __bf16* bp = Ut + (wv * 16 + l15) * UT_STR + kb + kh * 8;
      bt.q[0] = *(const v4u*)(bp);
      bt.q[1] = *(const v4u*)(bp + 16);
      #pragma unroll
      for (int mt = 0; mt < 4; ++mt) {
        Tile at;
        const __bf16* ap = hs + (mt * 16 + l15) * HS_STR + kb + kh * 8;
        at.q[0] = *(const v4u*)(ap);
        at.q[1] = *(const v4u*)(ap + 16);
        acc[mt] = __builtin_amdgcn_wmma_f32_16x16x32_bf16(
            false, at.v, false, bt.v, (short)0, acc[mt], false, false);
      }
    }

    __syncthreads();                            // xs reads done; zs may alias
    #pragma unroll
    for (int mt = 0; mt < 4; ++mt)
      #pragma unroll
      for (int r = 0; r < 8; ++r)
        zs[(mt * 16 + r + kh * 8) * ZS_STR + wv * 16 + l15] = acc[mt][r];
    __syncthreads();

    // ---- gates + masked state update + output write ----
    #pragma unroll
    for (int j = 0; j < 8; ++j) {
      const int b = bgI * 8 + j;
      const float* zr = zs + b * ZS_STR;
      float ig = sigmoidf_(zr[      ul] + bias_i);
      float fg = sigmoidf_(zr[ 32 + ul] + bias_f);
      float gg = tanhf    (zr[ 64 + ul] + bias_g);
      float og = sigmoidf_(zr[ 96 + ul] + bias_o);
      float cn = fg * creg[j] + ig * gg;
      float hn = og * tanhf(cn);
      if (msk[b]) { creg[j] = cn; hreg[j] = hn; }   // carry state where masked
      out[((size_t)b * T_ + t) * 1024 + dir * 512 + u0 + ul] = hreg[j];
      (hb_base + pw * (B_ * UNITS_))[b * UNITS_ + u0 + ul] = (__bf16)hreg[j];
    }

    // ---- per-direction grid barrier (fresh counter per step) ----
    __threadfence();
    __syncthreads();
    if (tid == 0) {
      __hip_atomic_fetch_add(&mycnt[s], 1, __ATOMIC_ACQ_REL,
                             __HIP_MEMORY_SCOPE_AGENT);
      while (__hip_atomic_load(&mycnt[s], __ATOMIC_ACQUIRE,
                               __HIP_MEMORY_SCOPE_AGENT) < NBLK_PER_DIR)
        __builtin_amdgcn_s_sleep(1);
    }
    __syncthreads();
  }

  // ---- final states ----
  #pragma unroll
  for (int j = 0; j < 8; ++j) {
    const int b = bgI * 8 + j;
    out[HC_OFF +        (size_t)b * 1024 + dir * 512 + u0 + ul] = hreg[j];
    out[HC_OFF + HSZ_ + (size_t)b * 1024 + dir * 512 + u0 + ul] = creg[j];
  }
}

extern "C" void kernel_launch(void* const* d_in, const int* in_sizes, int n_in,
                              void* d_out, int out_size, void* d_ws, size_t ws_size,
                              hipStream_t stream) {
  const int*   enc = (const int*)  d_in[0];
  const float* emb = (const float*)d_in[1];
  const float* Wf  = (const float*)d_in[2];
  const float* Uf  = (const float*)d_in[3];
  const float* bf  = (const float*)d_in[4];
  const float* Wb  = (const float*)d_in[5];
  const float* Ub  = (const float*)d_in[6];
  const float* bb  = (const float*)d_in[7];
  float* out = (float*)d_out;

  // workspace: [0,4096) step counters, then h double buffers (2 dir x 2 par)
  int*    cnt  = (int*)d_ws;
  __bf16* hbuf = (__bf16*)((char*)d_ws + 4096);
  const int ws_dwords = (4096 + 2 * 2 * B_ * UNITS_ * 2) / 4;   // 66560

  (void)hipFuncSetAttribute((const void*)lstm_bidir_kernel,
                            hipFuncAttributeMaxDynamicSharedMemorySize,
                            SMEM_BYTES);

  init_ws_kernel<<<64, 256, 0, stream>>>((unsigned*)d_ws, ws_dwords);
  lstm_bidir_kernel<<<dim3(2 * NBLK_PER_DIR), dim3(256), SMEM_BYTES, stream>>>(
      enc, emb, Wf, Uf, bf, Wb, Ub, bb, out, cnt, hbuf);
}
